// Multiheaded_GRUMix_Attention_10153302688114
// MI455X (gfx1250) — matrix operand order repaired and checked by reading, not compiled, not run on hardware
//
#include <hip/hip_runtime.h>
#include <hip/hip_bf16.h>
#include <math.h>

// ---------------------------------------------------------------------------
// Types & helpers
// ---------------------------------------------------------------------------
typedef __attribute__((ext_vector_type(16))) __bf16 v16bf;
typedef __attribute__((ext_vector_type(8)))  float  v8f;
typedef __attribute__((ext_vector_type(8)))  unsigned short u16x8;

__device__ __forceinline__ unsigned short f2bf(float f) {
  unsigned int x = __builtin_bit_cast(unsigned int, f);
  x += 0x7FFFu + ((x >> 16) & 1u);  // round-to-nearest-even
  return (unsigned short)(x >> 16);
}
__device__ __forceinline__ float bf2f(unsigned short h) {
  unsigned int x = ((unsigned int)h) << 16;
  return __builtin_bit_cast(float, x);
}
// fast hw ops: v_exp_f32 / v_rcp_f32 paths, no libm range-reduction chains
__device__ __forceinline__ float fexp(float x) { return __expf(x); }
__device__ __forceinline__ float frcp(float x) { return __builtin_amdgcn_rcpf(x); }
__device__ __forceinline__ float fsig(float x) { return frcp(1.f + fexp(-x)); }

__device__ __forceinline__ v8f v8zero() {
  v8f z;
#pragma unroll
  for (int i = 0; i < 8; i++) z[i] = 0.f;
  return z;
}
__device__ __forceinline__ v8f wmma_bf16(v16bf a, v16bf b, v8f c) {
  return __builtin_amdgcn_wmma_f32_16x16x32_bf16(false, a, false, b, (short)0, c,
                                                 false, false);
}
__device__ __forceinline__ float bfly_max16(float v) {
#pragma unroll
  for (int m = 1; m < 16; m <<= 1) v = fmaxf(v, __shfl_xor(v, m, 32));
  return v;
}
__device__ __forceinline__ float bfly_sum16(float v) {
#pragma unroll
  for (int m = 1; m < 16; m <<= 1) v += __shfl_xor(v, m, 32);
  return v;
}
__device__ __forceinline__ float bfly_sum32(float v) {
#pragma unroll
  for (int m = 1; m < 32; m <<= 1) v += __shfl_xor(v, m, 32);
  return v;
}

// A-fragment (16x32, M-major storage, row stride ld, ld % 8 == 0):
// lane -> row (lane&15); elems j -> K = kb+j, kb = (lane>=16)*16:
// 16 contiguous elements -> 2x 16B loads.
__device__ __forceinline__ v16bf load_frag_a(const unsigned short* p, int ld,
                                             int lane) {
  const unsigned short* r = p + (lane & 15) * ld + ((lane >> 4) << 4);
  union { v16bf v; u16x8 h[2]; } u;
  u.h[0] = *(const u16x8*)(r);
  u.h[1] = *(const u16x8*)(r + 8);
  return u.v;
}
// B-fragment (32x16) with LDS stored N-MAJOR: element (k,n) at p[n*ld + k],
// ld % 8 == 0. lane -> col (lane&15); elems j -> K = kb+j, kb = (lane>=16)*16:
// 16 contiguous elements -> 2x 16B loads.
__device__ __forceinline__ v16bf load_frag_b(const unsigned short* p, int ld,
                                             int lane) {
  const unsigned short* r = p + (lane & 15) * ld + ((lane >> 4) << 4);
  union { v16bf v; u16x8 h[2]; } u;
  u.h[0] = *(const u16x8*)(r);
  u.h[1] = *(const u16x8*)(r + 8);
  return u.v;
}

// ---------------------------------------------------------------------------
// f32 -> bf16 conversion
// ---------------------------------------------------------------------------
__global__ void k_f2bf(const float* __restrict__ s, unsigned short* __restrict__ d,
                       int n) {
  int i = blockIdx.x * blockDim.x + threadIdx.x;
  if (i < n) d[i] = f2bf(s[i]);
}

// ---------------------------------------------------------------------------
// Generic bf16 WMMA GEMM: C[M,N] = A[M,K] * B[K,N] (+bias) (+relu)
// BM=128, BN=64, BK=32; 256 threads = 8 waves (4 along M x 2 along N),
// each wave computes a 32x32 tile = 2x2 WMMA accumulators.
// B staged transposed (N-major) in LDS for contiguous fragment loads.
// ---------------------------------------------------------------------------
template <bool OBF, bool RELU, bool HASBIAS>
__global__ __launch_bounds__(256) void k_gemm_bf16(
    const unsigned short* __restrict__ A, int lda,
    const unsigned short* __restrict__ B, int ldb,
    const float* __restrict__ bias, void* __restrict__ C, int ldc, int M, int N,
    int K) {
  __shared__ alignas(16) unsigned short sA[128 * 40];   // [m][k], pad 8
  __shared__ alignas(16) unsigned short sBT[64 * 40];   // [n][k], pad 8
  const int tid  = threadIdx.x;
  const int lane = tid & 31;
  const int wave = tid >> 5;
  const int wm = wave & 3;   // 0..3 along M
  const int wn = wave >> 2;  // 0..1 along N
  const int m_blk = blockIdx.x * 128;
  const int n_blk = blockIdx.y * 64;

  v8f acc[2][2];
#pragma unroll
  for (int a = 0; a < 2; a++)
#pragma unroll
    for (int b = 0; b < 2; b++) acc[a][b] = v8zero();

  for (int k0 = 0; k0 < K; k0 += 32) {
    __syncthreads();
    for (int i = tid; i < 128 * 32; i += 256) {
      int r = i >> 5, c = i & 31;
      sA[r * 40 + c] = A[(size_t)(m_blk + r) * lda + k0 + c];
    }
    for (int i = tid; i < 32 * 64; i += 256) {
      int r = i >> 6, c = i & 63;  // r = k, c = n (coalesced global read)
      sBT[c * 40 + r] = B[(size_t)(k0 + r) * ldb + n_blk + c];
    }
    __syncthreads();
    v16bf af[2], bf_[2];
#pragma unroll
    for (int mi = 0; mi < 2; mi++)
      af[mi] = load_frag_a(sA + (wm * 32 + mi * 16) * 40, 40, lane);
#pragma unroll
    for (int ni = 0; ni < 2; ni++)
      bf_[ni] = load_frag_b(sBT + (wn * 32 + ni * 16) * 40, 40, lane);
#pragma unroll
    for (int mi = 0; mi < 2; mi++)
#pragma unroll
      for (int ni = 0; ni < 2; ni++)
        acc[mi][ni] = wmma_bf16(af[mi], bf_[ni], acc[mi][ni]);
  }

  const int cn = lane & 15, half = lane >> 4;
#pragma unroll
  for (int mi = 0; mi < 2; mi++)
#pragma unroll
    for (int ni = 0; ni < 2; ni++) {
      int mrow0 = m_blk + wm * 32 + mi * 16 + half * 8;
      int ncol  = n_blk + wn * 32 + ni * 16 + cn;
      float bval = HASBIAS ? bias[ncol] : 0.f;
#pragma unroll
      for (int i = 0; i < 8; i++) {
        float v = acc[mi][ni][i] + bval;
        if (RELU) v = fmaxf(v, 0.f);
        if (OBF)
          ((unsigned short*)C)[(size_t)(mrow0 + i) * ldc + ncol] = f2bf(v);
        else
          ((float*)C)[(size_t)(mrow0 + i) * ldc + ncol] = v;
      }
    }
}

// ---------------------------------------------------------------------------
// mix gate + mask lengths: one wave per (n,q) row (8192 rows)
// ---------------------------------------------------------------------------
__global__ __launch_bounds__(256) void k_mixlen(
    const float* __restrict__ Q, const float* __restrict__ mw,
    const float* __restrict__ mb, const int* __restrict__ mask,
    float* __restrict__ mix, float* __restrict__ step, float* __restrict__ len) {
  int r = (blockIdx.x * blockDim.x + threadIdx.x) >> 5;
  int lane = threadIdx.x & 31;
  float acc = 0.f, ls = 0.f;
  for (int c = lane; c < 512; c += 32) {
    acc += Q[(size_t)r * 512 + c] * mw[c];
    ls += (float)mask[(size_t)r * 512 + c];
  }
  acc = bfly_sum32(acc);
  ls  = bfly_sum32(ls);
  if (lane == 0) {
    mix[r]  = fsig(acc + mb[0]);
    len[r]  = ls;
    step[r] = frcp(fmaxf(1.f, ls - 1.f));
  }
}

// ---------------------------------------------------------------------------
// GRU gates + sigma/rho dots + mu: one wave per (n,h,q) row (65536 rows)
// ---------------------------------------------------------------------------
__global__ __launch_bounds__(256) void k_gate(
    const unsigned short* __restrict__ Gi, const unsigned short* __restrict__ Gh,
    const float* __restrict__ ps, const float* __restrict__ pa,
    const float* __restrict__ Wsig, const float* __restrict__ Wrho,
    const float* __restrict__ step, const float* __restrict__ len,
    float* __restrict__ mu, float* __restrict__ sigma) {
  int rs = (blockIdx.x * blockDim.x + threadIdx.x) >> 5;  // [N,H,q] order
  int lane = threadIdx.x & 31;
  int q = rs & 511, h = (rs >> 9) & 7, n = rs >> 12;
  int rx = ((n << 9) + q) * 8 + h;  // [n,q,h] order (Qpos layout)

  float psum = 0.f, r0 = 0.f, r1 = 0.f, r2 = 0.f;
#pragma unroll
  for (int e = 0; e < 2; e++) {
    int d = lane + e * 32;
    float ir  = bf2f(Gi[(size_t)rx * 192 + d]);
    float iz  = bf2f(Gi[(size_t)rx * 192 + 64 + d]);
    float in_ = bf2f(Gi[(size_t)rx * 192 + 128 + d]);
    float hr  = bf2f(Gh[(size_t)rs * 192 + d]);
    float hz  = bf2f(Gh[(size_t)rs * 192 + 64 + d]);
    float hn  = bf2f(Gh[(size_t)rs * 192 + 128 + d]);
    float hp  = ps[(size_t)rs * 64 + d];
    float r  = fsig(ir + hr);
    float z  = fsig(iz + hz);
    float nn = tanhf(in_ + r * hn);
    float st = (1.f - z) * nn + z * hp;
    psum += st * Wsig[h * 64 + d];
    r0 += st * Wrho[(h * 64 + d) * 3 + 0];
    r1 += st * Wrho[(h * 64 + d) * 3 + 1];
    r2 += st * Wrho[(h * 64 + d) * 3 + 2];
  }
  psum = bfly_sum32(psum);
  r0 = bfly_sum32(r0);
  r1 = bfly_sum32(r1);
  r2 = bfly_sum32(r2);
  if (lane == 0) {
    float L = len[(n << 9) + q];
    float sg = (fmaxf(psum, 0.f) + 0.27f) * frcp(L);
    float fl = floorf(r0);
    float steps = fl + fsig(10.f * (fabsf(r0 - fl) - 0.5f));
    float ag = fsig(r1), bg = fsig(r2);
    float mu_ = steps * step[(n << 9) + q] + ag * pa[rs] + bg;
    float muv = fmaxf(0.01f * mu_, fminf(mu_, 1.f + 0.01f * mu_));
    mu[rs] = muv;
    sigma[rs] = sg;
  }
}

// ---------------------------------------------------------------------------
// Attention pass 1: per-row softmax max/denominator + gaussian mass.
// grid = (QS/128, N*H); 8 waves x 16 q-rows; K chunks of 64 staged in LDS
// in natural [kcol][d] layout (B-fragment k-dim is d -> contiguous loads).
// ---------------------------------------------------------------------------
__global__ __launch_bounds__(256) void k_attn_stats(
    const unsigned short* __restrict__ Qh, const unsigned short* __restrict__ Kh,
    const int* __restrict__ mask, const float* __restrict__ vp,
    const float* __restrict__ step, const float* __restrict__ mu,
    const float* __restrict__ sigma, float* __restrict__ Mrow,
    float* __restrict__ Lrow, float* __restrict__ Sgrow) {
  __shared__ alignas(16) unsigned short sK[64 * 72];  // [kcol][d], pad 8
  const int lane = threadIdx.x & 31;
  const int wave = threadIdx.x >> 5;
  const int half = lane >> 4;
  const int cn = lane & 15;
  const int n = blockIdx.y >> 3;
  const int h = blockIdx.y & 7;
  const int q0 = blockIdx.x * 128 + wave * 16;

  const unsigned short* Qp = Qh + ((size_t)(n * 512 + q0)) * 512 + h * 64;
  v16bf qa0 = load_frag_a(Qp, 512, lane);
  v16bf qa1 = load_frag_a(Qp + 32, 512, lane);

  float M8[8], L8[8], G8[8], mu8[8], ni2s8[8], st8[8];
#pragma unroll
  for (int i = 0; i < 8; i++) {
    int q = q0 + half * 8 + i;
    int rs = ((n * 8 + h) << 9) + q;
    M8[i] = -__builtin_inff();
    L8[i] = 0.f;
    G8[i] = 0.f;
    mu8[i] = mu[rs];
    float s = sigma[rs];
    ni2s8[i] = -frcp(2.f * s * s);  // -1/(2*sigma^2), precomputed
    st8[i] = step[(n << 9) + q];
  }

  for (int kc = 0; kc < 8; kc++) {
    __syncthreads();
    for (int i = threadIdx.x; i < 4096; i += 256) {
      int d = i & 63, kcol = i >> 6;
      sK[kcol * 72 + d] =
          Kh[((size_t)(n * 512 + kc * 64 + kcol)) * 512 + h * 64 + d];
    }
    __syncthreads();
    float sarr[4][8], smax[8];
#pragma unroll
    for (int i = 0; i < 8; i++) smax[i] = -__builtin_inff();
#pragma unroll
    for (int t = 0; t < 4; t++) {
      v8f acc = v8zero();
      v16bf b0 = load_frag_b(sK + (t * 16) * 72, 72, lane);       // d = 0..31
      v16bf b1 = load_frag_b(sK + (t * 16) * 72 + 32, 72, lane);  // d = 32..63
      acc = wmma_bf16(qa0, b0, acc);
      acc = wmma_bf16(qa1, b1, acc);
      int k = kc * 64 + t * 16 + cn;
      float nv = vp[k];
      const int* mp = mask + ((size_t)n * 512 + (q0 + half * 8)) * 512 + k;
#pragma unroll
      for (int i = 0; i < 8; i++) {
        int mk = mp[(size_t)i * 512];
        float maskf = mk ? 1.f : 0.f;
        float sv = acc[i] * 0.125f;  // SCALING = 64^-0.5
        sv = mk ? sv : -__builtin_inff();
        sarr[t][i] = sv;
        smax[i] = fmaxf(smax[i], sv);
        float diff = nv * st8[i] - mu8[i];
        G8[i] += maskf * fexp(diff * diff * ni2s8[i]);
      }
    }
#pragma unroll
    for (int i = 0; i < 8; i++) {
      float cm = bfly_max16(smax[i]);
      float Mn = fmaxf(M8[i], cm);
      if (Mn > -1e30f) {
        float sc = (M8[i] > -1e30f) ? fexp(M8[i] - Mn) : 0.f;
        float es = 0.f;
#pragma unroll
        for (int t = 0; t < 4; t++) es += fexp(sarr[t][i] - Mn);
        L8[i] = L8[i] * sc + es;
        M8[i] = Mn;
      }
    }
  }
#pragma unroll
  for (int i = 0; i < 8; i++) {
    float Lt = bfly_sum16(L8[i]);
    float Gt = bfly_sum16(G8[i]);
    if (cn == 0) {
      int q = q0 + half * 8 + i;
      int rs = ((n * 8 + h) << 9) + q;
      Mrow[rs] = M8[i];
      Lrow[rs] = Lt;
      Sgrow[rs] = Gt + 512.f * 1e-20f;
    }
  }
}

// ---------------------------------------------------------------------------
// Attention pass 2: combined weights -> P*V via WMMA, write attended (bf16).
// K staged natural [kcol][d]; V staged transposed [v][kcol]; W tile per wave.
// ---------------------------------------------------------------------------
__global__ __launch_bounds__(256) void k_attn_apply(
    const unsigned short* __restrict__ Qh, const unsigned short* __restrict__ Kh,
    const unsigned short* __restrict__ Vh, const int* __restrict__ mask,
    const float* __restrict__ vp, const float* __restrict__ step,
    const float* __restrict__ mu, const float* __restrict__ sigma,
    const float* __restrict__ mixp, const float* __restrict__ Mrow,
    const float* __restrict__ Lrow, const float* __restrict__ Sgrow,
    unsigned short* __restrict__ attended) {
  __shared__ alignas(16) unsigned short sK[64 * 72];    // [kcol][d]
  __shared__ alignas(16) unsigned short sVT[64 * 72];   // [v][kcol]
  __shared__ alignas(16) unsigned short sW[8][16 * 64]; // per-wave [row][kcol]
  const int lane = threadIdx.x & 31;
  const int wave = threadIdx.x >> 5;
  const int half = lane >> 4;
  const int cn = lane & 15;
  const int n = blockIdx.y >> 3;
  const int h = blockIdx.y & 7;
  const int q0 = blockIdx.x * 128 + wave * 16;

  const unsigned short* Qp = Qh + ((size_t)(n * 512 + q0)) * 512 + h * 64;
  v16bf qa0 = load_frag_a(Qp, 512, lane);
  v16bf qa1 = load_frag_a(Qp + 32, 512, lane);

  float M8[8], iL8[8], gs8[8], mu8[8], ni2s8[8], st8[8];
#pragma unroll
  for (int i = 0; i < 8; i++) {
    int q = q0 + half * 8 + i;
    int rs = ((n * 8 + h) << 9) + q;
    float mx = mixp[(n << 9) + q];
    M8[i] = Mrow[rs];
    iL8[i] = (1.f - mx) * frcp(Lrow[rs]);
    gs8[i] = mx * frcp(Sgrow[rs]);
    mu8[i] = mu[rs];
    float s = sigma[rs];
    ni2s8[i] = -frcp(2.f * s * s);
    st8[i] = step[(n << 9) + q];
  }

  v8f out[4];
#pragma unroll
  for (int v = 0; v < 4; v++) out[v] = v8zero();

  for (int kc = 0; kc < 8; kc++) {
    __syncthreads();
    for (int i = threadIdx.x; i < 4096; i += 256) {
      int d = i & 63, kcol = i >> 6;
      size_t g = ((size_t)(n * 512 + kc * 64 + kcol)) * 512 + h * 64 + d;
      sK[kcol * 72 + d] = Kh[g];
      sVT[d * 72 + kcol] = Vh[g];
    }
    __syncthreads();
#pragma unroll
    for (int t = 0; t < 4; t++) {
      v8f acc = v8zero();
      v16bf b0 = load_frag_b(sK + (t * 16) * 72, 72, lane);
      v16bf b1 = load_frag_b(sK + (t * 16) * 72 + 32, 72, lane);
      acc = wmma_bf16(qa0, b0, acc);
      acc = wmma_bf16(qa1, b1, acc);
      int k = kc * 64 + t * 16 + cn;
      float nv = vp[k];
      const int* mp = mask + ((size_t)n * 512 + (q0 + half * 8)) * 512 + k;
#pragma unroll
      for (int i = 0; i < 8; i++) {
        int mk = mp[(size_t)i * 512];
        float maskf = mk ? 1.f : 0.f;
        float p = maskf * fexp(acc[i] * 0.125f - M8[i]);
        float diff = nv * st8[i] - mu8[i];
        float g = maskf * fexp(diff * diff * ni2s8[i]);
        float w = g * gs8[i] + p * iL8[i];
        sW[wave][(half * 8 + i) * 64 + t * 16 + cn] = f2bf(w);
      }
    }
    __syncthreads();
#pragma unroll
    for (int ks = 0; ks < 2; ks++) {
      v16bf wa = load_frag_a(&sW[wave][ks * 32], 64, lane);
#pragma unroll
      for (int vt = 0; vt < 4; vt++) {
        v16bf vb = load_frag_b(sVT + (vt * 16) * 72 + ks * 32, 72, lane);
        out[vt] = wmma_bf16(wa, vb, out[vt]);
      }
    }
  }
#pragma unroll
  for (int vt = 0; vt < 4; vt++)
#pragma unroll
    for (int i = 0; i < 8; i++) {
      int q = q0 + half * 8 + i;
      attended[((size_t)(n * 512 + q)) * 512 + h * 64 + vt * 16 + cn] =
          f2bf(out[vt][i]);
    }
}

// ---------------------------------------------------------------------------
// Host-side launch
// ---------------------------------------------------------------------------
extern "C" void kernel_launch(void* const* d_in, const int* in_sizes, int n_in,
                              void* d_out, int out_size, void* d_ws,
                              size_t ws_size, hipStream_t stream) {
  (void)in_sizes; (void)n_in; (void)out_size; (void)ws_size;
  const float* Q    = (const float*)d_in[0];
  const float* K    = (const float*)d_in[1];
  const float* V    = (const float*)d_in[2];
  const float* PS   = (const float*)d_in[3];
  const float* PA   = (const float*)d_in[4];
  const float* VP   = (const float*)d_in[5];
  const int*   MSK  = (const int*)d_in[6];
  const float* Wq   = (const float*)d_in[7];
  const float* Wk   = (const float*)d_in[8];
  const float* Wv   = (const float*)d_in[9];
  const float* Wpos = (const float*)d_in[10];
  const float* bpos = (const float*)d_in[11];
  const float* Wsig = (const float*)d_in[12];
  const float* Wrho = (const float*)d_in[13];
  const float* mw   = (const float*)d_in[14];
  const float* mb   = (const float*)d_in[15];
  const float* gwi  = (const float*)d_in[16];
  const float* gwh  = (const float*)d_in[17];
  const float* gbi  = (const float*)d_in[18];
  const float* gbh  = (const float*)d_in[19];
  const float* Wc   = (const float*)d_in[20];

  char* w = (char*)d_ws;
  size_t off = 0;
  auto take = [&](size_t bytes) -> void* {
    void* p = w + off;
    off = (off + bytes + 255) & ~(size_t)255;
    return p;
  };
  const size_t BIG = (size_t)8192 * 512;  // 4.19M elems
  unsigned short* Qb   = (unsigned short*)take(BIG * 2);
  unsigned short* Kb   = (unsigned short*)take(BIG * 2);
  unsigned short* Vb   = (unsigned short*)take(BIG * 2);
  unsigned short* PSb  = (unsigned short*)take(BIG * 2);
  unsigned short* Wqb  = (unsigned short*)take(512 * 512 * 2);
  unsigned short* Wkb  = (unsigned short*)take(512 * 512 * 2);
  unsigned short* Wvb  = (unsigned short*)take(512 * 512 * 2);
  unsigned short* Wpb  = (unsigned short*)take(512 * 512 * 2);
  unsigned short* Wcb  = (unsigned short*)take(512 * 512 * 2);
  unsigned short* gwib = (unsigned short*)take(64 * 192 * 2);
  unsigned short* gwhb = (unsigned short*)take(64 * 192 * 2);
  unsigned short* Qhb  = (unsigned short*)take(BIG * 2);
  unsigned short* Khb  = (unsigned short*)take(BIG * 2);
  unsigned short* Vhb  = (unsigned short*)take(BIG * 2);
  unsigned short* Xb   = (unsigned short*)take(BIG * 2);
  unsigned short* Gib  = (unsigned short*)take((size_t)65536 * 192 * 2);
  unsigned short* Ghb  = (unsigned short*)take((size_t)65536 * 192 * 2);
  unsigned short* Att  = (unsigned short*)take(BIG * 2);
  float* mixv = (float*)take(8192 * 4);
  float* stpv = (float*)take(8192 * 4);
  float* lenv = (float*)take(8192 * 4);
  float* muv  = (float*)take(65536 * 4);
  float* sgv  = (float*)take(65536 * 4);
  float* Mr   = (float*)take(65536 * 4);
  float* Lr   = (float*)take(65536 * 4);
  float* Sgr  = (float*)take(65536 * 4);

  auto cvt = [&](const float* s, unsigned short* d, int n) {
    k_f2bf<<<(n + 255) / 256, 256, 0, stream>>>(s, d, n);
  };
  cvt(Q, Qb, (int)BIG);
  cvt(K, Kb, (int)BIG);
  cvt(V, Vb, (int)BIG);
  cvt(PS, PSb, (int)BIG);
  cvt(Wq, Wqb, 512 * 512);
  cvt(Wk, Wkb, 512 * 512);
  cvt(Wv, Wvb, 512 * 512);
  cvt(Wpos, Wpb, 512 * 512);
  cvt(Wc, Wcb, 512 * 512);
  cvt(gwi, gwib, 64 * 192);
  cvt(gwh, gwhb, 64 * 192);

  // projections (bf16 out)
  {
    dim3 g(8192 / 128, 512 / 64);
    k_gemm_bf16<true, false, false><<<g, 256, 0, stream>>>(
        Qb, 512, Wqb, 512, nullptr, Qhb, 512, 8192, 512, 512);
    k_gemm_bf16<true, true, true><<<g, 256, 0, stream>>>(
        Qb, 512, Wpb, 512, bpos, Xb, 512, 8192, 512, 512);  // relu(Qpos)
    k_gemm_bf16<true, false, false><<<g, 256, 0, stream>>>(
        Kb, 512, Wkb, 512, nullptr, Khb, 512, 8192, 512, 512);
    k_gemm_bf16<true, false, false><<<g, 256, 0, stream>>>(
        Vb, 512, Wvb, 512, nullptr, Vhb, 512, 8192, 512, 512);
  }
  // GRU gate GEMMs: rows of 64, M = 16*512*8 = 65536
  {
    dim3 g(65536 / 128, 192 / 64);
    k_gemm_bf16<true, false, true><<<g, 256, 0, stream>>>(
        Xb, 64, gwib, 192, gbi, Gib, 192, 65536, 192, 64);
    k_gemm_bf16<true, false, true><<<g, 256, 0, stream>>>(
        PSb, 64, gwhb, 192, gbh, Ghb, 192, 65536, 192, 64);
  }

  k_mixlen<<<1024, 256, 0, stream>>>(Q, mw, mb, MSK, mixv, stpv, lenv);
  k_gate<<<8192, 256, 0, stream>>>(Gib, Ghb, PS, PA, Wsig, Wrho, stpv, lenv,
                                   muv, sgv);

  dim3 ag(4, 128);  // (QS/128, N*H)
  k_attn_stats<<<ag, 256, 0, stream>>>(Qhb, Khb, MSK, VP, stpv, muv, sgv, Mr,
                                       Lr, Sgr);
  k_attn_apply<<<ag, 256, 0, stream>>>(Qhb, Khb, Vhb, MSK, VP, stpv, muv, sgv,
                                       mixv, Mr, Lr, Sgr, Att);
  // out = attended @ Wc  (f32 output)
  {
    dim3 g(8192 / 128, 512 / 64);
    k_gemm_bf16<false, false, false><<<g, 256, 0, stream>>>(
        Att, 512, Wcb, 512, nullptr, d_out, 512, 8192, 512, 512);
  }
}
